// MultiHeadCrossAttention_26156350833219
// MI455X (gfx1250) — matrix operand order repaired and checked by reading, not compiled, not run on hardware
//
#include <hip/hip_runtime.h>

typedef __attribute__((ext_vector_type(16))) _Float16 v16h;
typedef __attribute__((ext_vector_type(8)))  _Float16 v8h;
typedef __attribute__((ext_vector_type(8)))  float    v8f;
typedef int v4i __attribute__((vector_size(16)));

#define E_DIM 1024
#define LQ    1024
#define LKV   2048
#define HDIM  64

enum { STORE_F16_ROW = 0, STORE_F32_ROW = 1, STORE_F16_TRANS = 2 };

// ---------------------------------------------------------------- fragments
__device__ __forceinline__ v16h mkfrag(const _Float16* p0, const _Float16* p1) {
    v8h a = *(const v8h*)p0;
    v8h b = *(const v8h*)p1;
    return __builtin_shufflevector(a, b, 0,1,2,3,4,5,6,7,8,9,10,11,12,13,14,15);
}

__device__ __forceinline__ v8f wmma_f16(v16h a, v16h b, v8f c) {
    return __builtin_amdgcn_wmma_f32_16x16x32_f16(false, a, false, b, (short)0, c, false, false);
}

// ------------------------------------------------- cross-lane (16) max, VALU
#if __has_builtin(__builtin_amdgcn_permlane16)
__device__ __forceinline__ float lane16_xor(float v, int lo, int hi) {
    int i = __float_as_int(v);
    return __int_as_float(__builtin_amdgcn_permlane16(i, i, lo, hi, false, false));
}
__device__ __forceinline__ float rowmax16(float t) {
    t = fmaxf(t, lane16_xor(t, (int)0x67452301, (int)0xEFCDAB89));  // xor 1
    t = fmaxf(t, lane16_xor(t, (int)0x54761032, (int)0xDCFE98BA));  // xor 2
    t = fmaxf(t, lane16_xor(t, (int)0x32107654, (int)0xBA98FEDC));  // xor 4
    t = fmaxf(t, lane16_xor(t, (int)0xFEDCBA98, (int)0x76543210));  // xor 8
    return t;
}
#else
__device__ __forceinline__ float rowmax16(float t) {
    t = fmaxf(t, __shfl_xor(t, 1, 32));
    t = fmaxf(t, __shfl_xor(t, 2, 32));
    t = fmaxf(t, __shfl_xor(t, 4, 32));
    t = fmaxf(t, __shfl_xor(t, 8, 32));
    return t;
}
#endif

// ------------------------------------------- global -> LDS 16-byte copy path
#if __has_builtin(__builtin_amdgcn_global_load_async_to_lds_b128)
#define HAVE_ASYNC_LDS 1
#endif

__device__ __forceinline__ void cp16_g2l(const _Float16* g, _Float16* l) {
#if defined(HAVE_ASYNC_LDS)
    __builtin_amdgcn_global_load_async_to_lds_b128(
        (__attribute__((address_space(1))) v4i*)(g),
        (__attribute__((address_space(3))) v4i*)(l), 0, 0);
#else
    *(v8h*)l = *(const v8h*)g;
#endif
}

__device__ __forceinline__ void wait_async_lds() {
#if defined(HAVE_ASYNC_LDS)
    asm volatile("s_wait_asynccnt 0" ::: "memory");
#endif
}

// ---------------------------------------------------------------- proj GEMM
// out = in @ W^T + bias  (K = N = 1024), 64x64 tile / 128-thread block.
template <typename InT, int MODE>
__global__ __launch_bounds__(128) void proj_gemm(const InT* __restrict__ in,
                                                 const float* __restrict__ W,
                                                 const float* __restrict__ bias,
                                                 void* __restrict__ outv,
                                                 float alpha, int transL) {
    __shared__ _Float16 sA[64][40];
    __shared__ _Float16 sB[64][40];

    const int tid  = threadIdx.x;
    const int wave = tid >> 5;
    const int lane = tid & 31;
    const int l15  = lane & 15;
    const int half = (lane < 16) ? 0 : 1;
    const int ntile = blockIdx.x;
    const int mtile = blockIdx.y;

    const int srow = tid >> 1;
    const int sseg = (tid & 1) * 16;
    const InT*   aptr = in + (size_t)(mtile * 64 + srow) * E_DIM + sseg;
    const float* bptr = W  + (size_t)(ntile * 64 + srow) * E_DIM + sseg;

    v8f acc[4] = {};

    for (int kk = 0; kk < E_DIM; kk += 32) {
        // stage A (convert to f16 when needed)
        if constexpr (sizeof(InT) == 4) {
            const float* g = (const float*)(aptr + kk);
#pragma unroll
            for (int i = 0; i < 16; i += 4) {
                float4 t = *(const float4*)(g + i);
                sA[srow][sseg + i + 0] = (_Float16)t.x;
                sA[srow][sseg + i + 1] = (_Float16)t.y;
                sA[srow][sseg + i + 2] = (_Float16)t.z;
                sA[srow][sseg + i + 3] = (_Float16)t.w;
            }
        } else {
            const v8h* g = (const v8h*)(aptr + kk);
            *(v8h*)&sA[srow][sseg]     = g[0];
            *(v8h*)&sA[srow][sseg + 8] = g[1];
        }
        { // stage B (weights, always f32)
            const float* g = bptr + kk;
#pragma unroll
            for (int i = 0; i < 16; i += 4) {
                float4 t = *(const float4*)(g + i);
                sB[srow][sseg + i + 0] = (_Float16)t.x;
                sB[srow][sseg + i + 1] = (_Float16)t.y;
                sB[srow][sseg + i + 2] = (_Float16)t.z;
                sB[srow][sseg + i + 3] = (_Float16)t.w;
            }
        }
        __syncthreads();

        const int ar  = wave * 16 + l15;
        const int k0a = half * 8;
        v16h afrag = mkfrag(&sA[ar][k0a], &sA[ar][k0a + 16]);
        const int k0b = half * 8;
#pragma unroll
        for (int j = 0; j < 4; ++j) {
            const int bc = j * 16 + l15;
            v16h bfrag = mkfrag(&sB[bc][k0b], &sB[bc][k0b + 16]);
            acc[j] = wmma_f16(afrag, bfrag, acc[j]);
        }
        __syncthreads();
    }

    const int gm = mtile * 64 + wave * 16 + half * 8;
#pragma unroll
    for (int j = 0; j < 4; ++j) {
        const int n  = ntile * 64 + j * 16 + l15;
        const float bv = bias[n];
        if constexpr (MODE == STORE_F16_ROW) {
            _Float16* out = (_Float16*)outv;
#pragma unroll
            for (int r = 0; r < 8; ++r)
                out[(size_t)(gm + r) * E_DIM + n] = (_Float16)((acc[j][r] + bv) * alpha);
        } else if constexpr (MODE == STORE_F32_ROW) {
            float* out = (float*)outv;
#pragma unroll
            for (int r = 0; r < 8; ++r)
                out[(size_t)(gm + r) * E_DIM + n] = (acc[j][r] + bv) * alpha;
        } else {
            _Float16* out = (_Float16*)outv;
            const int bidx = gm / transL;
            const int lr   = gm % transL;
            v8h hv;
#pragma unroll
            for (int r = 0; r < 8; ++r) hv[r] = (_Float16)((acc[j][r] + bv) * alpha);
            *(v8h*)(out + (size_t)bidx * E_DIM * transL + (size_t)n * transL + lr) = hv;
        }
    }
}

// ------------------------------------------------------- fused flash attention
// grid = (Lq/64, H, B), 128 threads (4 waves), 64-key chunks.
__global__ __launch_bounds__(128) void attn_fused(const _Float16* __restrict__ Qh,
                                                  const _Float16* __restrict__ Kh,
                                                  const _Float16* __restrict__ Vt,
                                                  _Float16* __restrict__ Ctx) {
    __shared__ _Float16 sK[64][72];       // [key][hd]
    __shared__ _Float16 sV[64][72];       // [d][kv]   (transposed V chunk)
    __shared__ _Float16 sP[4][16][72];    // per-wave P relayout buffer

    const int tid  = threadIdx.x;
    const int wave = tid >> 5;
    const int lane = tid & 31;
    const int l15  = lane & 15;
    const int half = (lane < 16) ? 0 : 1;
    const int qt = blockIdx.x, h = blockIdx.y, b = blockIdx.z;

    // Q fragments (A-operand layout) for this wave's 16 rows, in registers.
    const _Float16* qptr =
        Qh + (size_t)(b * LQ + qt * 64 + wave * 16 + l15) * E_DIM + h * HDIM;
    v16h qf[2];
#pragma unroll
    for (int s = 0; s < 2; ++s) {
        const int k0 = s * 32 + half * 8;
        qf[s] = mkfrag(qptr + k0, qptr + k0 + 16);
    }

    v16h ones;
#pragma unroll
    for (int i = 0; i < 16; ++i) ones[i] = (_Float16)1.0f;

    const _Float16* kbase = Kh + (size_t)(b * LKV) * E_DIM + h * HDIM;
    const _Float16* vbase = Vt + (size_t)b * E_DIM * LKV + (size_t)(h * HDIM) * LKV;

    float mrow[8], lrow[8], al[8];
#pragma unroll
    for (int r = 0; r < 8; ++r) { mrow[r] = -1e30f; lrow[r] = 0.0f; }
    v8f acc[4] = {};

    const int srow = tid >> 1;           // 0..63
    const int sseg = (tid & 1) * 32;     // 0 or 32

    for (int kv = 0; kv < LKV; kv += 64) {
        { // ---- cooperative staging: 64 keys of K, 64x64 of V^T (async -> LDS)
            const _Float16* kg = kbase + (size_t)(kv + srow) * E_DIM + sseg;
            const _Float16* vg = vbase + (size_t)srow * LKV + kv + sseg;
#pragma unroll
            for (int i = 0; i < 4; ++i) {
                cp16_g2l(kg + i * 8, &sK[srow][sseg + i * 8]);
                cp16_g2l(vg + i * 8, &sV[srow][sseg + i * 8]);
            }
            if (kv + 64 < LKV) {  // prefetch next chunk into cache hierarchy
                __builtin_prefetch(kg + 64 * E_DIM, 0, 1);
                __builtin_prefetch(vg + 64, 0, 1);
            }
        }
        wait_async_lds();
        __syncthreads();

        // ---- S = Q @ K^T : four 16x16 score tiles (64 keys), 8 WMMAs
        v8f st[4] = {};
#pragma unroll
        for (int s = 0; s < 2; ++s) {
            const int kb = s * 32 + half * 8;
#pragma unroll
            for (int j = 0; j < 4; ++j) {
                v16h bk = mkfrag(&sK[j * 16 + l15][kb], &sK[j * 16 + l15][kb + 16]);
                st[j] = wmma_f16(qf[s], bk, st[j]);
            }
        }

        // ---- online softmax: row max via permlane16 butterflies (VALU only)
#pragma unroll
        for (int r = 0; r < 8; ++r) {
            float t = fmaxf(fmaxf(st[0][r], st[1][r]), fmaxf(st[2][r], st[3][r]));
            t = rowmax16(t);
            const float mn = fmaxf(mrow[r], t);
            al[r]   = __expf(mrow[r] - mn);
            mrow[r] = mn;
            const int pr = half * 8 + r;
            sP[wave][pr][l15]      = (_Float16)__expf(st[0][r] - mn);
            sP[wave][pr][16 + l15] = (_Float16)__expf(st[1][r] - mn);
            sP[wave][pr][32 + l15] = (_Float16)__expf(st[2][r] - mn);
            sP[wave][pr][48 + l15] = (_Float16)__expf(st[3][r] - mn);
        }
        // cross-lane LDS RAW inside the wave: force DS completion
        asm volatile("s_wait_dscnt 0" ::: "memory");

        // P in A-operand layout (two k-steps of 32)
        const int k0p = half * 8;
        v16h ap0 = mkfrag(&sP[wave][l15][k0p],      &sP[wave][l15][k0p + 16]);
        v16h ap1 = mkfrag(&sP[wave][l15][32 + k0p], &sP[wave][l15][32 + k0p + 16]);

        // ---- row sums via WMMA: rowsum(P) = P @ ones  (replaces 32 shuffles)
        v8f rsum = {};
        rsum = wmma_f16(ap0, ones, rsum);
        rsum = wmma_f16(ap1, ones, rsum);

#pragma unroll
        for (int r = 0; r < 8; ++r) {
            lrow[r] = lrow[r] * al[r] + rsum[r];
            acc[0][r] *= al[r]; acc[1][r] *= al[r];
            acc[2][r] *= al[r]; acc[3][r] *= al[r];
        }

        // ---- O += P @ V : 8 WMMAs (d = 64, two k-steps)
#pragma unroll
        for (int j = 0; j < 4; ++j) {
            const int dc = j * 16 + l15;
            v16h bv0 = mkfrag(&sV[dc][half * 8],      &sV[dc][half * 8 + 16]);
            v16h bv1 = mkfrag(&sV[dc][32 + half * 8], &sV[dc][32 + half * 8 + 16]);
            acc[j] = wmma_f16(ap0, bv0, acc[j]);
            acc[j] = wmma_f16(ap1, bv1, acc[j]);
        }
        __syncthreads();
    }

    // ---- normalize, store ctx (f16 row-major [B*Lq, E], head-sliced columns)
    const int gq = b * LQ + qt * 64 + wave * 16 + half * 8;
#pragma unroll
    for (int j = 0; j < 4; ++j) {
        const int col = h * HDIM + j * 16 + l15;
#pragma unroll
        for (int r = 0; r < 8; ++r)
            Ctx[(size_t)(gq + r) * E_DIM + col] = (_Float16)(acc[j][r] / lrow[r]);
    }
}

// ------------------------------------------------------------------- launch
extern "C" void kernel_launch(void* const* d_in, const int* in_sizes, int n_in,
                              void* d_out, int out_size, void* d_ws, size_t ws_size,
                              hipStream_t stream) {
    (void)in_sizes; (void)n_in; (void)out_size; (void)ws_size;
    const float* x  = (const float*)d_in[0];
    const float* y  = (const float*)d_in[1];
    const float* Wq = (const float*)d_in[2];
    const float* bq = (const float*)d_in[3];
    const float* Wk = (const float*)d_in[4];
    const float* bk = (const float*)d_in[5];
    const float* Wv = (const float*)d_in[6];
    const float* bv = (const float*)d_in[7];
    const float* Wo = (const float*)d_in[8];
    const float* bo = (const float*)d_in[9];

    char* ws = (char*)d_ws;
    _Float16* Qh  = (_Float16*)(ws);                 //  8 MiB [B*Lq , E] f16 (scale folded)
    _Float16* Kh  = (_Float16*)(ws + (8u  << 20));   // 16 MiB [B*Lkv, E] f16
    _Float16* Vt  = (_Float16*)(ws + (24u << 20));   // 16 MiB [B, E, Lkv] f16 (transposed)
    _Float16* Ctx = (_Float16*)(ws + (40u << 20));   //  8 MiB [B*Lq , E] f16

    proj_gemm<float, STORE_F16_ROW><<<dim3(16, 64),  128, 0, stream>>>(x, Wq, bq, Qh, 0.125f, 0);
    proj_gemm<float, STORE_F16_ROW><<<dim3(16, 128), 128, 0, stream>>>(y, Wk, bk, Kh, 1.0f, 0);
    proj_gemm<float, STORE_F16_TRANS><<<dim3(16, 128), 128, 0, stream>>>(y, Wv, bv, Vt, 1.0f, LKV);
    attn_fused<<<dim3(LQ / 64, 16, 4), 128, 0, stream>>>(Qh, Kh, Vt, Ctx);
    proj_gemm<_Float16, STORE_F32_ROW><<<dim3(16, 64), 128, 0, stream>>>(Ctx, Wo, bo, d_out, 1.0f, 0);
}
